// MultiHeadAttention_1872605741669
// MI455X (gfx1250) — matrix operand order repaired and checked by reading, not compiled, not run on hardware
//
#include <hip/hip_runtime.h>
#include <hip/hip_bf16.h>
#include <stdint.h>

// ---------------------------------------------------------------------------
// Multi-head attention for MI455X (gfx1250), wave32 + WMMA f16 16x16x32.
//   B=4, L=2048, E=1024, H=16, D=64
// Pipeline:
//   cvt_f32_f16  x7 : x_q/x_k/x_v and Wq/Wk/Wv/Wo to f16 (one streaming pass)
//   gemm_f16     x3 : Q/K/V = x16 @ W16.T        (all-f16 WMMA GEMM, 32x64/wave)
//   attn_flash      : flash-attention, double-buffered LDS, K tile via TDM
//   out_proj        : out = ctx16 @ Wo16.T + bo  (f32 out)
// ---------------------------------------------------------------------------

typedef __attribute__((ext_vector_type(16))) _Float16 v16h;
typedef __attribute__((ext_vector_type(8)))  float    v8f;
typedef unsigned int u32x4 __attribute__((ext_vector_type(4)));
typedef int          i32x8 __attribute__((ext_vector_type(8)));
typedef int          i32x4 __attribute__((ext_vector_type(4)));

static constexpr int Bc = 4;
static constexpr int Lc = 2048;
static constexpr int Ec = 1024;
static constexpr int Hc = 16;
static constexpr int Dc = 64;
static constexpr int Mtot = Bc * Lc;          // 8192 rows in every GEMM

// --- Tensor Data Mover plumbing -------------------------------------------
// ROCm 7.2 (clang-22): 5-arg builtin, no TDM header.
// amdgpu-toolchain (clang-23 + therock headers): 6-arg builtin, header exists.
#if defined(__gfx1250__) && __has_builtin(__builtin_amdgcn_tensor_load_to_lds)
#define USE_TDM 1
#if __has_include(<hip/amd_detail/amd_gfx1250_TDM.h>)
#define TDM_6ARG 1
#endif
#else
#define USE_TDM 0
#endif

__device__ __forceinline__ void tdm_kload(const _Float16* gsrc, uint32_t lds_byte,
                                          const i32x8& g1) {
#if USE_TDM
    const uint64_t ga = (uint64_t)(uintptr_t)gsrc;
    u32x4 g0;
    g0[0] = 1u;                                               // count=1
    g0[1] = lds_byte;                                         // lds_addr
    g0[2] = (uint32_t)ga;                                     // global_addr lo
    g0[3] = (uint32_t)((ga >> 32) & 0x1FFFFFFu) | (2u << 30); // addr[56:32]+type=2
    const i32x4 z4 = {0, 0, 0, 0};
#ifdef TDM_6ARG
    const i32x8 z8 = {0, 0, 0, 0, 0, 0, 0, 0};
    __builtin_amdgcn_tensor_load_to_lds(g0, g1, z4, z4, z8, 0);
#else
    __builtin_amdgcn_tensor_load_to_lds(g0, g1, z4, z4, 0);
#endif
#endif
}

__device__ __forceinline__ void tdm_wait() {
#if USE_TDM && __has_builtin(__builtin_amdgcn_s_wait_tensorcnt)
    __builtin_amdgcn_s_wait_tensorcnt(0);
#endif
}

// --- WMMA helpers ----------------------------------------------------------
__device__ __forceinline__ v8f wmma16(v16h a, v16h b, v8f c) {
    return __builtin_amdgcn_wmma_f32_16x16x32_f16(
        false, a, false, b, (short)0, c, false, false);
}

union H2U { uint32_t u; _Float16 h[2]; };
union H8U { uint4 u; _Float16 h[8]; };

// A-fragment (16-bit A 16x32) k offset for VGPR v (halves 2v,2v+1)
__device__ __forceinline__ int a_koff(int v, int hi) {
    return (v < 4) ? (hi * 8 + 2 * v) : (16 + hi * 8 + 2 * (v - 4));
}
// B-fragment (16-bit B 32x16) k offset for VGPR v
__device__ __forceinline__ int b_koff(int v, int hi) {
    return (v < 4) ? (hi * 8 + 2 * v) : (16 + hi * 8 + 2 * (v - 4));
}

__device__ __forceinline__ void load_h2(const _Float16* p, v16h& f, int v) {
    H2U u; u.u = *reinterpret_cast<const uint32_t*>(p);
    f[2 * v] = u.h[0]; f[2 * v + 1] = u.h[1];
}

// ---------------------------------------------------------------------------
// Elementwise f32 -> f16 (8 elements / thread, b128-in / b128-out)
// ---------------------------------------------------------------------------
__global__ __launch_bounds__(256)
void cvt_f32_f16(const float* __restrict__ src, _Float16* __restrict__ dst) {
    const size_t i = ((size_t)blockIdx.x * blockDim.x + threadIdx.x) * 8;
    float4 f0 = *reinterpret_cast<const float4*>(src + i);
    float4 f1 = *reinterpret_cast<const float4*>(src + i + 4);
    H8U o;
    o.h[0] = (_Float16)f0.x; o.h[1] = (_Float16)f0.y;
    o.h[2] = (_Float16)f0.z; o.h[3] = (_Float16)f0.w;
    o.h[4] = (_Float16)f1.x; o.h[5] = (_Float16)f1.y;
    o.h[6] = (_Float16)f1.z; o.h[7] = (_Float16)f1.w;
    *reinterpret_cast<uint4*>(dst + i) = o.u;
}

// ---------------------------------------------------------------------------
// All-f16 GEMM: Y[m,n] = sum_k A[m,k] * W[n,k]   (f16 in, f16 out)
// Wave computes 32(M) x 64(N): 2 A-fragments shared across 4 B-fragments.
// ---------------------------------------------------------------------------
__global__ __launch_bounds__(256)
void gemm_f16(const _Float16* __restrict__ A, const _Float16* __restrict__ W,
              _Float16* __restrict__ Y) {
    const int lane = threadIdx.x & 31;
    const int wid  = blockIdx.x * 8 + (threadIdx.x >> 5);
    const int mt = wid >> 4;              // 256 tiles of 32 rows
    const int ng = wid & 15;              // 16 groups of 64 cols
    const int lo = lane & 15, hi = lane >> 4;

    const _Float16* arow0 = A + (size_t)(mt * 32 + lo) * Ec;
    const _Float16* arow1 = arow0 + (size_t)16 * Ec;
    const _Float16* wrow[4];
#pragma unroll
    for (int t = 0; t < 4; ++t)
        wrow[t] = W + (size_t)(ng * 64 + t * 16 + lo) * Ec;

    v8f acc[2][4];
#pragma unroll
    for (int r = 0; r < 2; ++r)
#pragma unroll
        for (int t = 0; t < 4; ++t)
#pragma unroll
            for (int i = 0; i < 8; ++i) acc[r][t][i] = 0.0f;

    for (int kk = 0; kk < Ec; kk += 32) {
        v16h a0, a1;
#pragma unroll
        for (int v = 0; v < 8; ++v) {
            load_h2(arow0 + kk + a_koff(v, hi), a0, v);
            load_h2(arow1 + kk + a_koff(v, hi), a1, v);
        }
#pragma unroll
        for (int t = 0; t < 4; ++t) {
            v16h b;
#pragma unroll
            for (int v = 0; v < 8; ++v) load_h2(wrow[t] + kk + b_koff(v, hi), b, v);
            acc[0][t] = wmma16(a0, b, acc[0][t]);
            acc[1][t] = wmma16(a1, b, acc[1][t]);
        }
    }
#pragma unroll
    for (int r = 0; r < 2; ++r)
#pragma unroll
        for (int t = 0; t < 4; ++t) {
            const int n = ng * 64 + t * 16 + lo;
#pragma unroll
            for (int v = 0; v < 8; ++v) {
                const int m = mt * 32 + r * 16 + v + hi * 8;
                Y[(size_t)m * Ec + n] = (_Float16)acc[r][t][v];
            }
        }
}

// ---------------------------------------------------------------------------
// Flash attention. Block = 8 waves sharing (b,h) and a 128-query slab.
// Double-buffered LDS staging: while tile kt is consumed (8 WMMA + softmax),
// tile kt+1 streams in (K via Tensor Data Mover, V via coalesced loads that
// are transposed on ds_store). One barrier per tile.
// ---------------------------------------------------------------------------
__global__ __launch_bounds__(256)
void attn_flash(const _Float16* __restrict__ Q16, const _Float16* __restrict__ K16,
                const _Float16* __restrict__ V16, _Float16* __restrict__ CTX) {
    __shared__ _Float16 sK[2][32 * 72];    // 32 keys x 64 d, row stride 72 halves
    __shared__ _Float16 sVT[2][64 * 40];   // 64 d x 32 keys (transposed), stride 40

    const int blk = blockIdx.x;
    const int qb  = blk & 15;
    const int bh  = blk >> 4;
    const int h   = bh & 15;
    const int b   = bh >> 4;

    const int lane = threadIdx.x & 31;
    const int wave = threadIdx.x >> 5;
    const int lo = lane & 15, hi = lane >> 4;
    const int qtile = qb * 8 + wave;
    const int hoff  = h * Dc;
    const int NT    = Lc / 32;

    // D# group1 (constant): data_size=2B, pad_enable, pad_interval=32 DW
    // (=128B tile row), pad_amount=4 DW (-> 144B LDS row stride), 2D tile
    // 64x32, tensor_dim0_stride = E.
    i32x8 g1;
    g1[0] = (1 << 16) | (1 << 20) | (4 << 22) | (3 << 25);
    g1[1] = (int)(64u << 16);          // tensor_dim0[15:0] = 64
    g1[2] = (int)(32u << 16);          // tensor_dim1[15:0] = 32
    g1[3] = (int)(64u << 16);          // tile_dim0 = 64
    g1[4] = 32;                        // tile_dim1 = 32, tile_dim2 = 0
    g1[5] = (int)Ec;                   // tensor_dim0_stride low 32
    g1[6] = 0;
    g1[7] = 0;
    const uint32_t ldsK[2] = { (uint32_t)(uintptr_t)(&sK[0][0]),
                               (uint32_t)(uintptr_t)(&sK[1][0]) };

    // Q as B-matrix of S^T = K * Q^T : column q, k-dim = d (two 32-wide steps)
    v16h bq[2];
    {
        const _Float16* qrow =
            Q16 + ((size_t)(b * Lc + qtile * 16 + lo)) * Ec + hoff;
#pragma unroll
        for (int g = 0; g < 2; ++g)
#pragma unroll
            for (int v = 0; v < 8; ++v)
                load_h2(qrow + g * 32 + b_koff(v, hi), bq[g], v);
    }

    v8f O[4];
#pragma unroll
    for (int g = 0; g < 4; ++g)
#pragma unroll
        for (int i = 0; i < 8; ++i) O[g][i] = 0.0f;
    float mrun = -1.0e30f, srun = 0.0f;

    const int trow = threadIdx.x >> 3;          // 0..31 (key row)
    const int tc8  = (threadIdx.x & 7) * 8;     // 0..56 (d chunk of 8)
    const size_t rowoff = (size_t)(b * Lc + trow) * Ec + hoff + tc8;
    const _Float16* kbase = K16 + (size_t)b * Lc * Ec + hoff;   // tile row 0

    // ---- prologue: stage tile 0 into buffer 0 ----
    {
        if (wave == 0) tdm_kload(kbase, ldsK[0], g1);
#if !USE_TDM
        *reinterpret_cast<uint4*>(&sK[0][trow * 72 + tc8]) =
            *reinterpret_cast<const uint4*>(K16 + rowoff);
#endif
        H8U cv; cv.u = *reinterpret_cast<const uint4*>(V16 + rowoff);
#pragma unroll
        for (int j = 0; j < 8; ++j) sVT[0][(tc8 + j) * 40 + trow] = cv.h[j];
        if (wave == 0) tdm_wait();
    }
    __syncthreads();

    for (int kt = 0; kt < NT; ++kt) {
        const int buf = kt & 1;
        const bool have_next = (kt + 1) < NT;

        // ---- kick off next tile's traffic before computing this one ----
        H8U cvn;
#if !USE_TDM
        uint4 kn;
#endif
        if (have_next) {
            const size_t nrow = rowoff + (size_t)(kt + 1) * 32 * Ec;
            cvn.u = *reinterpret_cast<const uint4*>(V16 + nrow);
#if USE_TDM
            if (wave == 0)
                tdm_kload(kbase + (size_t)(kt + 1) * 32 * Ec, ldsK[buf ^ 1], g1);
#else
            kn = *reinterpret_cast<const uint4*>(K16 + nrow);
#endif
        }

        // ---- compute on current buffer ----
        const _Float16* kb = &sK[buf][0];
        const _Float16* vb = &sVT[buf][0];

        v8f c[2];
#pragma unroll
        for (int s = 0; s < 2; ++s) {
#pragma unroll
            for (int i = 0; i < 8; ++i) c[s][i] = 0.0f;
            const int key = s * 16 + lo;        // A row (lane-indexed)
#pragma unroll
            for (int g = 0; g < 2; ++g) {
                v16h ak;
#pragma unroll
                for (int v = 0; v < 8; ++v)
                    load_h2(&kb[key * 72 + g * 32 + a_koff(v, hi)], ak, v);
                c[s] = wmma16(ak, bq[g], c[s]);
            }
        }
        // scale 1/sqrt(D) and online softmax over this tile's 32 keys
        float tmax = -1.0e30f;
#pragma unroll
        for (int s = 0; s < 2; ++s)
#pragma unroll
            for (int i = 0; i < 8; ++i) {
                c[s][i] *= 0.125f;
                tmax = fmaxf(tmax, c[s][i]);
            }
        tmax = fmaxf(tmax, __shfl_xor(tmax, 16, 32));
        const float mnew  = fmaxf(mrun, tmax);
        const float alpha = __expf(mrun - mnew);
        mrun = mnew;

        float p0[8], p1[8], tsum = 0.0f;
#pragma unroll
        for (int i = 0; i < 8; ++i) {
            p0[i] = __expf(c[0][i] - mnew);
            p1[i] = __expf(c[1][i] - mnew);
            tsum += p0[i] + p1[i];
        }
        tsum += __shfl_xor(tsum, 16, 32);
        srun = srun * alpha + tsum;

        // P^T C-fragment -> P A-fragment: lane-local repack
        v16h aP;
#pragma unroll
        for (int i = 0; i < 8; ++i) {
            aP[i]     = (_Float16)p0[i];
            aP[8 + i] = (_Float16)p1[i];
        }
        // per-row rescale (row m = v + 8*hi; stats live in lane m)
        float av[8];
#pragma unroll
        for (int v = 0; v < 8; ++v) av[v] = __shfl(alpha, v + hi * 8, 32);

#pragma unroll
        for (int g = 0; g < 4; ++g) {
            v16h bv;
            const int dl = g * 16 + lo;         // B column = d
#pragma unroll
            for (int v = 0; v < 8; ++v)
                load_h2(&vb[dl * 40 + b_koff(v, hi)], bv, v);
#pragma unroll
            for (int i = 0; i < 8; ++i) O[g][i] *= av[i];
            O[g] = wmma16(aP, bv, O[g]);
        }

        // ---- drain next tile's staging, single barrier per tile ----
        if (have_next) {
#pragma unroll
            for (int j = 0; j < 8; ++j)
                sVT[buf ^ 1][(tc8 + j) * 40 + trow] = cvn.h[j];
#if !USE_TDM
            *reinterpret_cast<uint4*>(&sK[buf ^ 1][trow * 72 + tc8]) = kn;
#endif
            if (wave == 0) tdm_wait();
        }
        __syncthreads();
    }

    // normalize and store ctx (f16)
    const float inv = 1.0f / srun;
    float iv[8];
#pragma unroll
    for (int v = 0; v < 8; ++v) iv[v] = __shfl(inv, v + hi * 8, 32);
#pragma unroll
    for (int g = 0; g < 4; ++g) {
        const int d = hoff + g * 16 + lo;
#pragma unroll
        for (int v = 0; v < 8; ++v) {
            const int q = qtile * 16 + v + hi * 8;
            CTX[(size_t)(b * Lc + q) * Ec + d] = (_Float16)(O[g][v] * iv[v]);
        }
    }
}

// ---------------------------------------------------------------------------
// Output projection: out[m,n] = sum_k ctx16[m,k]*Wo16[n,k] + bo[n]  (f32 out)
// ---------------------------------------------------------------------------
__global__ __launch_bounds__(256)
void out_proj(const _Float16* __restrict__ A, const _Float16* __restrict__ W,
              const float* __restrict__ bias, float* __restrict__ Y) {
    const int lane = threadIdx.x & 31;
    const int wid  = blockIdx.x * 8 + (threadIdx.x >> 5);
    const int mt = wid >> 4;
    const int ng = wid & 15;
    const int lo = lane & 15, hi = lane >> 4;

    const _Float16* arow0 = A + (size_t)(mt * 32 + lo) * Ec;
    const _Float16* arow1 = arow0 + (size_t)16 * Ec;
    const _Float16* wrow[4];
    float bn[4];
#pragma unroll
    for (int t = 0; t < 4; ++t) {
        const int n = ng * 64 + t * 16 + lo;
        wrow[t] = W + (size_t)n * Ec;
        bn[t]   = bias[n];
    }

    v8f acc[2][4];
#pragma unroll
    for (int r = 0; r < 2; ++r)
#pragma unroll
        for (int t = 0; t < 4; ++t)
#pragma unroll
            for (int i = 0; i < 8; ++i) acc[r][t][i] = 0.0f;

    for (int kk = 0; kk < Ec; kk += 32) {
        v16h a0, a1;
#pragma unroll
        for (int v = 0; v < 8; ++v) {
            load_h2(arow0 + kk + a_koff(v, hi), a0, v);
            load_h2(arow1 + kk + a_koff(v, hi), a1, v);
        }
#pragma unroll
        for (int t = 0; t < 4; ++t) {
            v16h b;
#pragma unroll
            for (int v = 0; v < 8; ++v) load_h2(wrow[t] + kk + b_koff(v, hi), b, v);
            acc[0][t] = wmma16(a0, b, acc[0][t]);
            acc[1][t] = wmma16(a1, b, acc[1][t]);
        }
    }
#pragma unroll
    for (int r = 0; r < 2; ++r)
#pragma unroll
        for (int t = 0; t < 4; ++t) {
            const int n = ng * 64 + t * 16 + lo;
#pragma unroll
            for (int v = 0; v < 8; ++v) {
                const int m = mt * 32 + r * 16 + v + hi * 8;
                Y[(size_t)m * Ec + n] = acc[r][t][v] + bn[t];
            }
        }
}

// ---------------------------------------------------------------------------
extern "C" void kernel_launch(void* const* d_in, const int* in_sizes, int n_in,
                              void* d_out, int out_size, void* d_ws, size_t ws_size,
                              hipStream_t stream) {
    const float* x_q = (const float*)d_in[0];
    const float* x_k = (const float*)d_in[1];
    const float* x_v = (const float*)d_in[2];
    const float* Wq  = (const float*)d_in[3];
    const float* Wk  = (const float*)d_in[4];
    const float* Wv  = (const float*)d_in[5];
    const float* Wo  = (const float*)d_in[6];
    const float* bo  = (const float*)d_in[7];
    float* out = (float*)d_out;

    const size_t SZ = (size_t)Bc * Lc * Ec;   // 8 Mi elements (act tensors)
    const size_t WZ = (size_t)Ec * Ec;        // 1 Mi elements (weights)
    _Float16* base = (_Float16*)d_ws;
    _Float16* Xq16 = base;                    // later reused as CTX
    _Float16* Xk16 = base + SZ;
    _Float16* Xv16 = base + 2 * SZ;
    _Float16* Wq16 = base + 3 * SZ;
    _Float16* Wk16 = Wq16 + WZ;
    _Float16* Wv16 = Wk16 + WZ;
    _Float16* Wo16 = Wv16 + WZ;
    _Float16* Q16  = Wo16 + WZ;
    _Float16* K16  = Q16 + SZ;
    _Float16* V16  = K16 + SZ;
    _Float16* CTX  = Xq16;                    // alias: x_q16 dead after proj Q

    const int cvtA = (int)(SZ / 8 / 256);     // 4096 blocks (activations)
    const int cvtW = (int)(WZ / 8 / 256);     // 512 blocks  (weights)
    const int gemm_blocks = (Mtot / 32) * (Ec / 64) / 8;   // 512
    const int attn_blocks = Bc * Hc * (Lc / 128);          // 1024

    cvt_f32_f16<<<cvtA, 256, 0, stream>>>(x_q, Xq16);
    cvt_f32_f16<<<cvtA, 256, 0, stream>>>(x_k, Xk16);
    cvt_f32_f16<<<cvtA, 256, 0, stream>>>(x_v, Xv16);
    cvt_f32_f16<<<cvtW, 256, 0, stream>>>(Wq, Wq16);
    cvt_f32_f16<<<cvtW, 256, 0, stream>>>(Wk, Wk16);
    cvt_f32_f16<<<cvtW, 256, 0, stream>>>(Wv, Wv16);
    cvt_f32_f16<<<cvtW, 256, 0, stream>>>(Wo, Wo16);

    gemm_f16<<<gemm_blocks, 256, 0, stream>>>(Xq16, Wq16, Q16);
    gemm_f16<<<gemm_blocks, 256, 0, stream>>>(Xk16, Wk16, K16);
    gemm_f16<<<gemm_blocks, 256, 0, stream>>>(Xv16, Wv16, V16);

    attn_flash<<<attn_blocks, 256, 0, stream>>>(Q16, K16, V16, CTX);

    out_proj<<<gemm_blocks, 256, 0, stream>>>(CTX, Wo16, bo, out);
}
